// HGT_44203803411104
// MI455X (gfx1250) — compile-verified, matrix operand order checked by reading, not run on hardware
//
#include <hip/hip_runtime.h>
#include <math.h>

// ---------------------------------------------------------------------------
// HGT forward for MI455X (gfx1250, wave32).
// Dense node GEMMs -> v_wmma_f32_16x16x32_bf16 (f32 accumulate).
// Edge phase -> float4 gathers, wave shuffles, f32 global atomics.
// ---------------------------------------------------------------------------

typedef __attribute__((ext_vector_type(16))) __bf16 v16bf;
typedef __attribute__((ext_vector_type(8)))  float  v8f;

__device__ __forceinline__ unsigned short f2bf(float f) {
  unsigned u = __float_as_uint(f);
  u += 0x7FFFu + ((u >> 16) & 1u);        // round-to-nearest-even
  return (unsigned short)(u >> 16);
}

__device__ __forceinline__ float gelu_exact(float x) {
  return 0.5f * x * (1.0f + erff(x * 0.70710678118654752f));
}

__device__ __forceinline__ void atomicMaxFloat(float* addr, float val) {
  if (val >= 0.0f) atomicMax((int*)addr, __float_as_int(val));
  else             atomicMin((unsigned int*)addr, (unsigned int)__float_as_int(val));
}

__global__ void hgt_fill_f32(float* __restrict__ p, float v, long n) {
  long i = (long)blockIdx.x * blockDim.x + threadIdx.x;
  if (i < n) p[i] = v;
}

// ---------------------------------------------------------------------------
// Y[N x 128] = EPI( PRE(X) @ W + bias )
//  PRE:   0 = none, 1 = exact gelu on X
//  EPI:   0 = none, 1 = relu, 2 = sigmoid-skip mix with xprev then relu
//  WMODE: 0 = dense W[128][128] (k-major), 1 = block-diag from W[H][32][32]
// Block: 256 threads = 8 waves; each block computes a 128x128 output tile.
// ---------------------------------------------------------------------------
template <int PRE, int EPI, int WMODE>
__global__ __launch_bounds__(256)
void gemm128_wmma(const float* __restrict__ X, const float* __restrict__ W,
                  const float* __restrict__ bias,
                  const float* __restrict__ xprev,
                  const float* __restrict__ skipg,
                  float* __restrict__ Y, int N) {
  __shared__ unsigned short As[128 * 136];   // bf16 activations, row-major
  __shared__ unsigned short Bs[128 * 136];   // bf16 W^T: Bs[n][k]
  const int tid = threadIdx.x;
  const int rowBase = blockIdx.x * 128;

  // stage weights transposed (Bs[n][k] = W[k][n]) as bf16
  for (int t = tid; t < 128 * 32; t += 256) {
    int k  = t >> 5;
    int n4 = (t & 31) << 2;
    float4 w;
    if (WMODE == 0) {
      w = *reinterpret_cast<const float4*>(W + k * 128 + n4);
    } else {
      int hk = k >> 5;
      if (hk == (n4 >> 5))
        w = *reinterpret_cast<const float4*>(W + hk * 1024 + (k & 31) * 32 + (n4 & 31));
      else
        w = make_float4(0.f, 0.f, 0.f, 0.f);
    }
    Bs[(n4 + 0) * 136 + k] = f2bf(w.x);
    Bs[(n4 + 1) * 136 + k] = f2bf(w.y);
    Bs[(n4 + 2) * 136 + k] = f2bf(w.z);
    Bs[(n4 + 3) * 136 + k] = f2bf(w.w);
  }
  // stage activations (optionally gelu'd) as bf16
  for (int t = tid; t < 128 * 32; t += 256) {
    int r  = t >> 5;
    int c4 = (t & 31) << 2;
    int g  = rowBase + r;
    float4 x = make_float4(0.f, 0.f, 0.f, 0.f);
    if (g < N) x = *reinterpret_cast<const float4*>(X + (long)g * 128 + c4);
    if (PRE == 1) {
      x.x = gelu_exact(x.x); x.y = gelu_exact(x.y);
      x.z = gelu_exact(x.z); x.w = gelu_exact(x.w);
    }
    As[r * 136 + c4 + 0] = f2bf(x.x);
    As[r * 136 + c4 + 1] = f2bf(x.y);
    As[r * 136 + c4 + 2] = f2bf(x.z);
    As[r * 136 + c4 + 3] = f2bf(x.w);
  }
  __syncthreads();

  const int wave = tid >> 5;
  const int lane = tid & 31;
  const int half = lane >> 4;       // K-half select per ISA bf16 A layout
  const int lr   = lane & 15;       // M (for A) / N (for B) index
  const int m    = wave * 16 + lr;

  v8f acc[8];
#pragma unroll
  for (int i = 0; i < 8; ++i)
    acc[i] = (v8f){0.f, 0.f, 0.f, 0.f, 0.f, 0.f, 0.f, 0.f};

#pragma unroll
  for (int kk = 0; kk < 4; ++kk) {
    const int kbase = kk * 32;
    union { unsigned u[8]; v16bf v; } A;
#pragma unroll
    for (int j = 0; j < 8; ++j) {
      // CDNA5 16-bit A 16x32 layout: VGPR j holds packed K pair
      int k = kbase + 2 * j + (half ? 8 : 0) + (j >= 4 ? 8 : 0);
      A.u[j] = *reinterpret_cast<const unsigned*>(&As[m * 136 + k]);
    }
#pragma unroll
    for (int ct = 0; ct < 8; ++ct) {
      union { unsigned u[8]; v16bf v; } B;
      int n = ct * 16 + lr;
#pragma unroll
      for (int j = 0; j < 8; ++j) {
        int k = kbase + 2 * j + (half ? 8 : 0) + (j >= 4 ? 8 : 0);
        B.u[j] = *reinterpret_cast<const unsigned*>(&Bs[n * 136 + k]);
      }
      acc[ct] = __builtin_amdgcn_wmma_f32_16x16x32_bf16(
          false, A.v, false, B.v, (short)0, acc[ct], false, false);
    }
  }

  float aG = 0.f;
  if (EPI == 2) aG = 1.f / (1.f + expf(-skipg[0]));

#pragma unroll
  for (int ct = 0; ct < 8; ++ct) {
    const int col = ct * 16 + lr;
#pragma unroll
    for (int g = 0; g < 8; ++g) {
      int row = rowBase + wave * 16 + g + 8 * half;  // C/D layout: M = vgpr + 8*half
      if (row < N) {
        float v = acc[ct][g];
        if (bias) v += bias[col];
        if (EPI == 1) v = fmaxf(v, 0.f);
        if (EPI == 2) {
          float p = xprev[(long)row * 128 + col];
          v = aG * v + (1.f - aG) * p;
          v = fmaxf(v, 0.f);
        }
        Y[(long)row * 128 + col] = v;
      }
    }
  }
}

// ---------------------------------------------------------------------------
// Edge phase. One wave per edge: lane l -> head h = l>>3, 4 channels each.
// ---------------------------------------------------------------------------
__global__ __launch_bounds__(256)
void edge_logits(const float* __restrict__ krel, const float* __restrict__ q,
                 const int* __restrict__ src, const int* __restrict__ dst,
                 const float* __restrict__ prel, float* __restrict__ dmax,
                 float* __restrict__ logits, int E, float scale) {
  int e = blockIdx.x * 8 + (threadIdx.x >> 5);
  if (e >= E) return;
  int lane = threadIdx.x & 31;
  int h    = lane >> 3;
  int off  = (lane & 7) << 2;
  int s = src[e], d = dst[e];
  float4 kv = *reinterpret_cast<const float4*>(krel + (long)s * 128 + h * 32 + off);
  float4 qv = *reinterpret_cast<const float4*>(q    + (long)d * 128 + h * 32 + off);
  float p = kv.x * qv.x + kv.y * qv.y + kv.z * qv.z + kv.w * qv.w;
  p += __shfl_xor(p, 1, 32);
  p += __shfl_xor(p, 2, 32);
  p += __shfl_xor(p, 4, 32);
  if ((lane & 7) == 0) {
    float lg = p * prel[h] * scale;
    logits[(long)e * 4 + h] = lg;
    atomicMaxFloat(&dmax[(long)d * 4 + h], lg);
  }
}

__global__ void edge_exp(float* __restrict__ logits, const int* __restrict__ dst,
                         const float* __restrict__ dmax, float* __restrict__ denom,
                         int EH) {
  int i = blockIdx.x * blockDim.x + threadIdx.x;
  if (i >= EH) return;
  int e = i >> 2, h = i & 3;
  int d = dst[e];
  float ev = expf(logits[i] - dmax[(long)d * 4 + h]);
  logits[i] = ev;
  atomicAdd(&denom[(long)d * 4 + h], ev);
}

__global__ __launch_bounds__(256)
void edge_scatter(const float* __restrict__ mrel, const float* __restrict__ evals,
                  const int* __restrict__ src, const int* __restrict__ dst,
                  const float* __restrict__ denom, float* __restrict__ agg, int E) {
  int e = blockIdx.x * 8 + (threadIdx.x >> 5);
  if (e >= E) return;
  int lane = threadIdx.x & 31;
  int h    = lane >> 3;
  int off  = (lane & 7) << 2;
  int s = src[e], d = dst[e];
  float alpha = evals[(long)e * 4 + h] / (denom[(long)d * 4 + h] + 1e-16f);
  float4 mv = *reinterpret_cast<const float4*>(mrel + (long)s * 128 + h * 32 + off);
  float* o = agg + (long)d * 128 + h * 32 + off;
  atomicAdd(o + 0, mv.x * alpha);
  atomicAdd(o + 1, mv.y * alpha);
  atomicAdd(o + 2, mv.z * alpha);
  atomicAdd(o + 3, mv.w * alpha);
}

// ---------------------------------------------------------------------------
extern "C" void kernel_launch(void* const* d_in, const int* in_sizes, int n_in,
                              void* d_out, int out_size, void* d_ws, size_t ws_size,
                              hipStream_t stream) {
  (void)in_sizes; (void)n_in; (void)out_size; (void)ws_size;
  const int NU = 200000, NN = 100000, C = 128, H = 4, E = 250000, L = 2;
  const float scale = 1.0f / sqrtf(32.0f);

  const float* x_user = (const float*)d_in[0];
  const float* x_news = (const float*)d_in[1];
  const int* srcs[3] = {(const int*)d_in[2], (const int*)d_in[4], (const int*)d_in[6]};
  const int* dsts[3] = {(const int*)d_in[3], (const int*)d_in[5], (const int*)d_in[7]};
  const float* lin_W = (const float*)d_in[8];
  const float* lin_b = (const float*)d_in[9];
  const float* kW = (const float*)d_in[10];
  const float* kb = (const float*)d_in[11];
  const float* qW = (const float*)d_in[12];
  const float* qb = (const float*)d_in[13];
  const float* vW = (const float*)d_in[14];
  const float* vb = (const float*)d_in[15];
  const float* aW = (const float*)d_in[16];
  const float* ab = (const float*)d_in[17];
  const float* skip = (const float*)d_in[18];
  const float* arel = (const float*)d_in[19];
  const float* mrel = (const float*)d_in[20];
  const float* prel = (const float*)d_in[21];
  float* out = (float*)d_out;

  float* w = (float*)d_ws;
  size_t off = 0;
  auto alloc = [&](size_t n) { float* p = w + off; off += (n + 3) & ~(size_t)3; return p; };

  float* xs[2]  = {alloc((size_t)NU * C), alloc((size_t)NN * C)};
  float* kq[2]  = {alloc((size_t)NU * C), alloc((size_t)NN * C)};
  float* qq[2]  = {alloc((size_t)NU * C), alloc((size_t)NN * C)};
  float* vq[2]  = {alloc((size_t)NU * C), alloc((size_t)NN * C)};
  float* krelb  = alloc((size_t)NU * C);
  float* mrelb  = alloc((size_t)NU * C);
  float* agg[2] = {alloc((size_t)NU * C), alloc((size_t)NN * C)};
  float* logitsE = alloc((size_t)E * H);
  float* dmax    = alloc((size_t)NU * H);
  float* denom   = alloc((size_t)NU * H);

  const int Ns[2] = {NU, NN};
  const int stv[3] = {0, 1, 0}, dtv[3] = {1, 0, 0};  // (user->news),(news->user),(user->user)

  dim3 blk(256);
  auto gblk = [](int N) { return dim3((unsigned)((N + 127) / 128)); };

  // input projection + relu
  gemm128_wmma<0, 1, 0><<<gblk(NU), blk, 0, stream>>>(x_user, lin_W, lin_b,
                                                      nullptr, nullptr, xs[0], NU);
  gemm128_wmma<0, 1, 0><<<gblk(NN), blk, 0, stream>>>(x_news, lin_W + C * C, lin_b + C,
                                                      nullptr, nullptr, xs[1], NN);

  for (int l = 0; l < L; ++l) {
    for (int t = 0; t < 2; ++t) {
      long wo = (long)(l * 2 + t) * C * C;
      long bo = (long)(l * 2 + t) * C;
      gemm128_wmma<0, 0, 0><<<gblk(Ns[t]), blk, 0, stream>>>(xs[t], kW + wo, kb + bo,
                                                             nullptr, nullptr, kq[t], Ns[t]);
      gemm128_wmma<0, 0, 0><<<gblk(Ns[t]), blk, 0, stream>>>(xs[t], qW + wo, qb + bo,
                                                             nullptr, nullptr, qq[t], Ns[t]);
      gemm128_wmma<0, 0, 0><<<gblk(Ns[t]), blk, 0, stream>>>(xs[t], vW + wo, vb + bo,
                                                             nullptr, nullptr, vq[t], Ns[t]);
    }
    for (int t = 0; t < 2; ++t) {
      long n = (long)Ns[t] * C;
      hgt_fill_f32<<<dim3((unsigned)((n + 255) / 256)), blk, 0, stream>>>(agg[t], 0.f, n);
    }
    for (int r = 0; r < 3; ++r) {
      int st = stv[r], dt = dtv[r];
      long ro = (long)(l * 3 + r) * H * 32 * 32;
      // per-head relation transforms as block-diagonal WMMA GEMMs
      gemm128_wmma<0, 0, 1><<<gblk(Ns[st]), blk, 0, stream>>>(kq[st], arel + ro, nullptr,
                                                              nullptr, nullptr, krelb, Ns[st]);
      gemm128_wmma<0, 0, 1><<<gblk(Ns[st]), blk, 0, stream>>>(vq[st], mrel + ro, nullptr,
                                                              nullptr, nullptr, mrelb, Ns[st]);
      long nh = (long)Ns[dt] * H;
      hgt_fill_f32<<<dim3((unsigned)((nh + 255) / 256)), blk, 0, stream>>>(dmax, -INFINITY, nh);
      hgt_fill_f32<<<dim3((unsigned)((nh + 255) / 256)), blk, 0, stream>>>(denom, 0.f, nh);
      edge_logits<<<dim3((unsigned)((E + 7) / 8)), blk, 0, stream>>>(
          krelb, qq[dt], srcs[r], dsts[r], prel + (long)(l * 3 + r) * H,
          dmax, logitsE, E, scale);
      edge_exp<<<dim3((unsigned)((E * H + 255) / 256)), blk, 0, stream>>>(
          logitsE, dsts[r], dmax, denom, E * H);
      edge_scatter<<<dim3((unsigned)((E + 7) / 8)), blk, 0, stream>>>(
          mrelb, logitsE, srcs[r], dsts[r], denom, agg[dt], E);
    }
    for (int t = 0; t < 2; ++t) {
      long wo = (long)(l * 2 + t) * C * C;
      long bo = (long)(l * 2 + t) * C;
      float* y = (l == L - 1) ? (t == 0 ? out : out + (long)NU * C) : xs[t];
      // gelu(agg) @ aW + ab, sigmoid-skip with xs, relu  (in-place safe: elementwise in xprev)
      gemm128_wmma<1, 2, 0><<<gblk(Ns[t]), blk, 0, stream>>>(agg[t], aW + wo, ab + bo,
                                                             xs[t], skip + l * 2 + t, y, Ns[t]);
    }
  }
}
